// SEGNNModel_23905787969631
// MI455X (gfx1250) — matrix operand order repaired and checked by reading, not compiled, not run on hardware
//
#include <hip/hip_runtime.h>

// ---------------------------------------------------------------------------
// SEGNN forward for MI455X (gfx1250, wave32).
// Big GEMMs run on v_wmma_f32_16x16x32_bf16 (bf16 in, fp32 accumulate);
// gates / CG paths / biases / atomics in fp32 VALU.
// ---------------------------------------------------------------------------

constexpr int NNODES = 32768;
constexpr int NEDGES = 262144;
constexpr float RSQRT3 = 0.5773502691896258f;

typedef __bf16 bf16x16 __attribute__((ext_vector_type(16)));
typedef float v8f __attribute__((ext_vector_type(8)));

__device__ __forceinline__ __bf16 to_bf(float f) {
  unsigned int u = __float_as_uint(f);
  u += 0x7FFFu + ((u >> 16) & 1u);            // round-to-nearest-even
  unsigned short s = (unsigned short)(u >> 16);
  return __builtin_bit_cast(__bf16, s);
}

__device__ __forceinline__ float sigm(float x) {
  return 1.0f / (1.0f + __expf(-x));
}

union AB {
  bf16x16 v;
  struct { uint4 lo, hi; } q;
};
static_assert(sizeof(AB) == 32, "AB frag size");

// GEMM: D[16 rows][NT*16 cols] = A[16][KPAD] (bf16, LDS) x Bt[N][KPAD] (bf16,
// global, row n = output channel) accumulated fp32, + optional bias, stored to
// an LDS fp32 buffer with stride `ostride`.
// A fragment (16-bit A 16x32 layout): lane(m=l&15, h=l>>4), elems 0..7 ->
// K=kb+8h+e, elems 8..15 -> K=kb+16+8h+(e-8) -> two contiguous 16B chunks.
// B fragment: lane n=l&15, same K mapping as A -> two contiguous 16B chunks.
// D: elem r -> row r+8h, col nt*16+(l&15).
template <int KPAD, int NT>
__device__ __forceinline__ void gemm_store(const __bf16* __restrict__ A,
                                           const __bf16* __restrict__ Bt,
                                           float* __restrict__ outp, int ostride,
                                           const float* __restrict__ bias,
                                           int lane) {
  const int m = lane & 15;
  const int half = lane >> 4;
#pragma unroll
  for (int nt = 0; nt < NT; ++nt) {
    v8f acc = {};
#pragma unroll
    for (int kb = 0; kb < KPAD; kb += 32) {
      AB a, b;
      const __bf16* a0 = A + m * KPAD + kb + half * 8;
      a.q.lo = *(const uint4*)(a0);
      a.q.hi = *(const uint4*)(a0 + 16);
      const __bf16* bp = Bt + (nt * 16 + m) * KPAD + kb + half * 8;
      b.q.lo = *(const uint4*)(bp);
      b.q.hi = *(const uint4*)(bp + 16);
      acc = __builtin_amdgcn_wmma_f32_16x16x32_bf16(false, a.v, false, b.v,
                                                    (short)0, acc, false, false);
    }
    const int col = nt * 16 + m;
    const float bv = bias ? bias[col] : 0.0f;
#pragma unroll
    for (int r = 0; r < 8; ++r)
      outp[(r + 8 * half) * ostride + col] = acc[r] + bv;
  }
}

// Pack weights: dst[n][k] (bf16, Kpad padded) = concat_k(A[Ka][N], B[Kb][N])[k][n]
__global__ void pack_w(__bf16* __restrict__ dst, const float* __restrict__ A,
                       const float* __restrict__ B, int Ka, int Kb, int N,
                       int Kpad) {
  int idx = blockIdx.x * blockDim.x + threadIdx.x;
  if (idx >= N * Kpad) return;
  int n = idx / Kpad, k = idx - n * Kpad;
  float v = 0.0f;
  if (k < Ka) v = A[k * N + n];
  else if (k < Ka + Kb) v = B[(k - Ka) * N + n];
  dst[n * Kpad + k] = to_bf(v);
}

// ---------------------------------------------------------------------------
// Edge kernel: one wave (32 threads) per 16-edge tile.
// m1: K(s)=129+64(vdot)->224, N=96 ; sv: K=129->160, N=32 ; vv: K=64, N=32 x3
// m2: K(s)=64+32->96, N=96       ; sv: K=64, N=32      ; vv: K=32, N=32 x3
// ---------------------------------------------------------------------------
__global__ __launch_bounds__(32) void edge_kernel(
    const float* __restrict__ x_s, const float* __restrict__ x_v,
    const int* __restrict__ eidx, const float* __restrict__ edist,
    const float* __restrict__ eas, const float* __restrict__ eav,
    const __bf16* __restrict__ W1p, const __bf16* __restrict__ W1svT,
    const __bf16* __restrict__ W1vvT, const float* __restrict__ b1,
    const __bf16* __restrict__ W2p, const __bf16* __restrict__ W2svT,
    const __bf16* __restrict__ W2vvT, const float* __restrict__ b2,
    float* __restrict__ agg_s, float* __restrict__ agg_v,
    float* __restrict__ sum_as, float* __restrict__ sum_av,
    float* __restrict__ cntb) {
  __shared__ __align__(16) __bf16 F[16 * 224];   // m1 scalar feats, reused K=96 for m2
  __shared__ __align__(16) __bf16 G[16 * 160];   // unscaled s_cat, reused K=64
  __shared__ __align__(16) __bf16 Vb[3 * 16 * 64];  // v_c*a_s, reused [3][16][32]
  __shared__ float OUTS[16 * 96];
  __shared__ float SWb[16 * 32];
  __shared__ float VWb[3 * 16 * 32];
  __shared__ float AS[16];
  __shared__ float AV[16 * 3];
  __shared__ int DN[16];

  const int lane = threadIdx.x;
  const int ebase = blockIdx.x * 16;

  // ---- stage 1: gather node features, build m1 GEMM operands (bf16) -------
  {
    const int i = lane >> 1, hc = lane & 1;
    const int e = ebase + i;
    const int sn = eidx[e];
    const int dn = eidx[NEDGES + e];
    const float as = eas[e];
    const float av0 = eav[e * 3 + 0], av1 = eav[e * 3 + 1], av2 = eav[e * 3 + 2];
    if (hc == 0) {
      const float dv = edist[e];
      F[i * 224 + 128] = to_bf(dv * as);
      G[i * 160 + 128] = to_bf(dv);
      for (int c = 129; c < 160; ++c) G[i * 160 + c] = to_bf(0.0f);
      for (int c = 193; c < 224; ++c) F[i * 224 + c] = to_bf(0.0f);
      AS[i] = as;
      AV[i * 3 + 0] = av0; AV[i * 3 + 1] = av1; AV[i * 3 + 2] = av2;
      DN[i] = dn;
    }
    for (int c = 0; c < 64; ++c) {            // s_cat = [x_s[dst], x_s[src]]
      const int col = hc * 64 + c;
      const float v = (col < 64) ? x_s[(size_t)dn * 64 + col]
                                 : x_s[(size_t)sn * 64 + col - 64];
      F[i * 224 + col] = to_bf(v * as);
      G[i * 160 + col] = to_bf(v);
    }
    for (int q = 0; q < 32; ++q) {            // v_cat = [x_v[dst], x_v[src]]
      const int vi = hc * 32 + q;
      const int node = (vi < 32) ? dn : sn;
      const float* vp = x_v + ((size_t)node * 32 + (vi & 31)) * 3;
      const float v0 = vp[0], v1 = vp[1], v2 = vp[2];
      F[i * 224 + 129 + vi] = to_bf((v0 * av0 + v1 * av1 + v2 * av2) * RSQRT3);
      Vb[0 * 1024 + i * 64 + vi] = to_bf(v0 * as);
      Vb[1 * 1024 + i * 64 + vi] = to_bf(v1 * as);
      Vb[2 * 1024 + i * 64 + vi] = to_bf(v2 * as);
    }
  }
  __syncthreads();
  // ---- m1 GEMMs -----------------------------------------------------------
  gemm_store<224, 6>(F, W1p, OUTS, 96, b1, lane);
  gemm_store<160, 2>(G, W1svT, SWb, 32, nullptr, lane);
  gemm_store<64, 2>(Vb + 0, W1vvT, VWb + 0, 32, nullptr, lane);
  gemm_store<64, 2>(Vb + 1024, W1vvT, VWb + 512, 32, nullptr, lane);
  gemm_store<64, 2>(Vb + 2048, W1vvT, VWb + 1024, 32, nullptr, lane);
  __syncthreads();
  // ---- gate m1, build m2 operands ----------------------------------------
  {
    const int i = lane & 15, rep = lane >> 4;
    const float as = AS[i];
    const float av0 = AV[i * 3], av1 = AV[i * 3 + 1], av2 = AV[i * 3 + 2];
    for (int h = rep * 32; h < rep * 32 + 32; ++h) {
      const float x = OUTS[i * 96 + h];
      const float ms = x * sigm(x);
      F[i * 96 + h] = to_bf(ms * as);
      G[i * 64 + h] = to_bf(ms);
    }
    for (int o = rep * 16; o < rep * 16 + 16; ++o) {
      const float gate = sigm(OUTS[i * 96 + 64 + o]);
      const float sw = SWb[i * 32 + o];
      const float m0 = gate * (sw * av0 + VWb[0 * 512 + i * 32 + o]);
      const float m1 = gate * (sw * av1 + VWb[1 * 512 + i * 32 + o]);
      const float m2 = gate * (sw * av2 + VWb[2 * 512 + i * 32 + o]);
      F[i * 96 + 64 + o] = to_bf((m0 * av0 + m1 * av1 + m2 * av2) * RSQRT3);
      Vb[0 * 512 + i * 32 + o] = to_bf(m0 * as);
      Vb[1 * 512 + i * 32 + o] = to_bf(m1 * as);
      Vb[2 * 512 + i * 32 + o] = to_bf(m2 * as);
    }
  }
  __syncthreads();
  // ---- m2 GEMMs -----------------------------------------------------------
  gemm_store<96, 6>(F, W2p, OUTS, 96, b2, lane);
  gemm_store<64, 2>(G, W2svT, SWb, 32, nullptr, lane);
  gemm_store<32, 2>(Vb + 0, W2vvT, VWb + 0, 32, nullptr, lane);
  gemm_store<32, 2>(Vb + 512, W2vvT, VWb + 512, 32, nullptr, lane);
  gemm_store<32, 2>(Vb + 1024, W2vvT, VWb + 1024, 32, nullptr, lane);
  __syncthreads();
  // ---- gate m2, scatter-add to dst nodes ---------------------------------
  {
    const int i = lane & 15, rep = lane >> 4;
    const int dn = DN[i];
    const float as = AS[i];
    const float av0 = AV[i * 3], av1 = AV[i * 3 + 1], av2 = AV[i * 3 + 2];
    for (int h = rep * 32; h < rep * 32 + 32; ++h) {
      const float x = OUTS[i * 96 + h];
      unsafeAtomicAdd(&agg_s[(size_t)dn * 64 + h], x * sigm(x));
    }
    for (int o = rep * 16; o < rep * 16 + 16; ++o) {
      const float gate = sigm(OUTS[i * 96 + 64 + o]);
      const float sw = SWb[i * 32 + o];
      float* vout = &agg_v[((size_t)dn * 32 + o) * 3];
      unsafeAtomicAdd(vout + 0, gate * (sw * av0 + VWb[0 * 512 + i * 32 + o]));
      unsafeAtomicAdd(vout + 1, gate * (sw * av1 + VWb[1 * 512 + i * 32 + o]));
      unsafeAtomicAdd(vout + 2, gate * (sw * av2 + VWb[2 * 512 + i * 32 + o]));
    }
    if (rep == 0) {                           // per-edge attr aggregation
      unsafeAtomicAdd(&sum_as[dn], as);
      unsafeAtomicAdd(&sum_av[dn * 3 + 0], av0);
      unsafeAtomicAdd(&sum_av[dn * 3 + 1], av1);
      unsafeAtomicAdd(&sum_av[dn * 3 + 2], av2);
      unsafeAtomicAdd(&cntb[dn], 1.0f);
    }
  }
}

// ---------------------------------------------------------------------------
// Node kernel: one wave per 16-node tile.
// u1: K(s)=128+64->192, N=96 ; sv: K=128, N=32 ; vv: K=64, N=32 x3 ; gated
// u2: K(s)=64+32->96,  N=64 ; sv: K=64,  N=32 ; vv: K=32, N=32 x3 ; no gate
// ---------------------------------------------------------------------------
__global__ __launch_bounds__(32) void node_kernel(
    const float* __restrict__ x_s, const float* __restrict__ x_v,
    const float* __restrict__ agg_s, const float* __restrict__ agg_v,
    const float* __restrict__ sum_as, const float* __restrict__ sum_av,
    const float* __restrict__ cntb,
    const __bf16* __restrict__ U1p, const __bf16* __restrict__ U1svT,
    const __bf16* __restrict__ U1vvT, const float* __restrict__ ub1,
    const __bf16* __restrict__ U2p, const __bf16* __restrict__ U2svT,
    const __bf16* __restrict__ U2vvT, const float* __restrict__ ub2,
    float* __restrict__ out) {
  __shared__ __align__(16) __bf16 F[16 * 192];
  __shared__ __align__(16) __bf16 G[16 * 128];
  __shared__ __align__(16) __bf16 Vb[3 * 16 * 64];
  __shared__ float OUTS[16 * 96];
  __shared__ float SWb[16 * 32];
  __shared__ float VWb[3 * 16 * 32];
  __shared__ float NAS[16];
  __shared__ float NAV[16 * 3];

  const int lane = threadIdx.x;
  const int nbase = blockIdx.x * 16;

  if (lane < 16) {
    const int node = nbase + lane;
    float c = cntb[node];
    c = (c < 1.0f) ? 1.0f : c;
    NAS[lane] = sum_as[node] / c;
    NAV[lane * 3 + 0] = sum_av[node * 3 + 0] / c;
    NAV[lane * 3 + 1] = sum_av[node * 3 + 1] / c;
    NAV[lane * 3 + 2] = sum_av[node * 3 + 2] / c;
  }
  __syncthreads();
  // ---- stage features for u1 ---------------------------------------------
  {
    const int i = lane >> 1, hc = lane & 1;
    const int node = nbase + i;
    const float nas = NAS[i];
    const float nv0 = NAV[i * 3], nv1 = NAV[i * 3 + 1], nv2 = NAV[i * 3 + 2];
    for (int c = 0; c < 64; ++c) {            // s_cat = [x_s, agg_s]
      const int col = hc * 64 + c;
      const float v = (col < 64) ? x_s[(size_t)node * 64 + col]
                                 : agg_s[(size_t)node * 64 + col - 64];
      F[i * 192 + col] = to_bf(v * nas);
      G[i * 128 + col] = to_bf(v);
    }
    for (int q = 0; q < 32; ++q) {            // v_cat = [x_v, agg_v]
      const int vi = hc * 32 + q;
      const float* vp = (vi < 32) ? x_v + ((size_t)node * 32 + vi) * 3
                                  : agg_v + ((size_t)node * 32 + vi - 32) * 3;
      const float v0 = vp[0], v1 = vp[1], v2 = vp[2];
      F[i * 192 + 128 + vi] = to_bf((v0 * nv0 + v1 * nv1 + v2 * nv2) * RSQRT3);
      Vb[0 * 1024 + i * 64 + vi] = to_bf(v0 * nas);
      Vb[1 * 1024 + i * 64 + vi] = to_bf(v1 * nas);
      Vb[2 * 1024 + i * 64 + vi] = to_bf(v2 * nas);
    }
  }
  __syncthreads();
  gemm_store<192, 6>(F, U1p, OUTS, 96, ub1, lane);
  gemm_store<128, 2>(G, U1svT, SWb, 32, nullptr, lane);
  gemm_store<64, 2>(Vb + 0, U1vvT, VWb + 0, 32, nullptr, lane);
  gemm_store<64, 2>(Vb + 1024, U1vvT, VWb + 512, 32, nullptr, lane);
  gemm_store<64, 2>(Vb + 2048, U1vvT, VWb + 1024, 32, nullptr, lane);
  __syncthreads();
  // ---- gate u1, build u2 operands ----------------------------------------
  {
    const int i = lane & 15, rep = lane >> 4;
    const float nas = NAS[i];
    const float nv0 = NAV[i * 3], nv1 = NAV[i * 3 + 1], nv2 = NAV[i * 3 + 2];
    for (int h = rep * 32; h < rep * 32 + 32; ++h) {
      const float x = OUTS[i * 96 + h];
      const float us = x * sigm(x);
      F[i * 96 + h] = to_bf(us * nas);
      G[i * 64 + h] = to_bf(us);
    }
    for (int o = rep * 16; o < rep * 16 + 16; ++o) {
      const float gate = sigm(OUTS[i * 96 + 64 + o]);
      const float sw = SWb[i * 32 + o];
      const float u0 = gate * (sw * nv0 + VWb[0 * 512 + i * 32 + o]);
      const float u1 = gate * (sw * nv1 + VWb[1 * 512 + i * 32 + o]);
      const float u2 = gate * (sw * nv2 + VWb[2 * 512 + i * 32 + o]);
      F[i * 96 + 64 + o] = to_bf((u0 * nv0 + u1 * nv1 + u2 * nv2) * RSQRT3);
      Vb[0 * 512 + i * 32 + o] = to_bf(u0 * nas);
      Vb[1 * 512 + i * 32 + o] = to_bf(u1 * nas);
      Vb[2 * 512 + i * 32 + o] = to_bf(u2 * nas);
    }
  }
  __syncthreads();
  gemm_store<96, 4>(F, U2p, OUTS, 96, ub2, lane);   // N=64 -> cols 0..63
  gemm_store<64, 2>(G, U2svT, SWb, 32, nullptr, lane);
  gemm_store<32, 2>(Vb + 0, U2vvT, VWb + 0, 32, nullptr, lane);
  gemm_store<32, 2>(Vb + 512, U2vvT, VWb + 512, 32, nullptr, lane);
  gemm_store<32, 2>(Vb + 1024, U2vvT, VWb + 1024, 32, nullptr, lane);
  __syncthreads();
  // ---- residual add + write output [N,160] -------------------------------
  {
    const int i = lane & 15, rep = lane >> 4;
    const int node = nbase + i;
    const float nv0 = NAV[i * 3], nv1 = NAV[i * 3 + 1], nv2 = NAV[i * 3 + 2];
    for (int h = rep * 32; h < rep * 32 + 32; ++h)
      out[(size_t)node * 160 + h] = x_s[(size_t)node * 64 + h] + OUTS[i * 96 + h];
    for (int o = rep * 16; o < rep * 16 + 16; ++o) {
      const float sw = SWb[i * 32 + o];
      const size_t ob = (size_t)node * 160 + 64 + o * 3;
      const size_t vb = ((size_t)node * 32 + o) * 3;
      out[ob + 0] = x_v[vb + 0] + sw * nv0 + VWb[0 * 512 + i * 32 + o];
      out[ob + 1] = x_v[vb + 1] + sw * nv1 + VWb[1 * 512 + i * 32 + o];
      out[ob + 2] = x_v[vb + 2] + sw * nv2 + VWb[2 * 512 + i * 32 + o];
    }
  }
}

// ---------------------------------------------------------------------------
extern "C" void kernel_launch(void* const* d_in, const int* in_sizes, int n_in,
                              void* d_out, int out_size, void* d_ws,
                              size_t ws_size, hipStream_t stream) {
  (void)in_sizes; (void)n_in; (void)out_size; (void)ws_size;
  const float* x_s = (const float*)d_in[0];
  const float* x_v = (const float*)d_in[1];
  const int* eidx = (const int*)d_in[2];
  const float* edist = (const float*)d_in[3];
  const float* eas = (const float*)d_in[4];
  const float* eav = (const float*)d_in[5];
  const float* m1_ss = (const float*)d_in[6];
  const float* m1_vs = (const float*)d_in[7];
  const float* m1_sv = (const float*)d_in[8];
  const float* m1_vv = (const float*)d_in[9];
  const float* m1_b = (const float*)d_in[10];
  const float* m2_ss = (const float*)d_in[11];
  const float* m2_vs = (const float*)d_in[12];
  const float* m2_sv = (const float*)d_in[13];
  const float* m2_vv = (const float*)d_in[14];
  const float* m2_b = (const float*)d_in[15];
  const float* u1_ss = (const float*)d_in[16];
  const float* u1_vs = (const float*)d_in[17];
  const float* u1_sv = (const float*)d_in[18];
  const float* u1_vv = (const float*)d_in[19];
  const float* u1_b = (const float*)d_in[20];
  const float* u2_ss = (const float*)d_in[21];
  const float* u2_vs = (const float*)d_in[22];
  const float* u2_sv = (const float*)d_in[23];
  const float* u2_vv = (const float*)d_in[24];
  const float* u2_b = (const float*)d_in[25];
  float* out = (float*)d_out;

  char* ws = (char*)d_ws;
  size_t off = 0;
  auto take = [&](size_t bytes) -> char* {
    char* p = ws + off;
    off = (off + bytes + 255) & ~(size_t)255;
    return p;
  };
  // Zero-initialized aggregation buffers (contiguous, memset below).
  float* agg_s = (float*)take((size_t)NNODES * 64 * 4);
  float* agg_v = (float*)take((size_t)NNODES * 96 * 4);
  float* sum_as = (float*)take((size_t)NNODES * 4);
  float* sum_av = (float*)take((size_t)NNODES * 3 * 4);
  float* cntb = (float*)take((size_t)NNODES * 4);
  const size_t zbytes = off;
  // Packed bf16 weights: [N][Kpad], transposed/padded.
  __bf16* W1p = (__bf16*)take(96 * 224 * 2);
  __bf16* W1svT = (__bf16*)take(32 * 160 * 2);
  __bf16* W1vvT = (__bf16*)take(32 * 64 * 2);
  __bf16* W2p = (__bf16*)take(96 * 96 * 2);
  __bf16* W2svT = (__bf16*)take(32 * 64 * 2);
  __bf16* W2vvT = (__bf16*)take(32 * 32 * 2);
  __bf16* U1p = (__bf16*)take(96 * 192 * 2);
  __bf16* U1svT = (__bf16*)take(32 * 128 * 2);
  __bf16* U1vvT = (__bf16*)take(32 * 64 * 2);
  __bf16* U2p = (__bf16*)take(64 * 96 * 2);
  __bf16* U2svT = (__bf16*)take(32 * 64 * 2);
  __bf16* U2vvT = (__bf16*)take(32 * 32 * 2);

  hipMemsetAsync(d_ws, 0, zbytes, stream);

  auto cdiv = [](int a, int b) { return (a + b - 1) / b; };
  pack_w<<<cdiv(96 * 224, 256), 256, 0, stream>>>(W1p, m1_ss, m1_vs, 129, 64, 96, 224);
  pack_w<<<cdiv(32 * 160, 256), 256, 0, stream>>>(W1svT, m1_sv, nullptr, 129, 0, 32, 160);
  pack_w<<<cdiv(32 * 64, 256), 256, 0, stream>>>(W1vvT, m1_vv, nullptr, 64, 0, 32, 64);
  pack_w<<<cdiv(96 * 96, 256), 256, 0, stream>>>(W2p, m2_ss, m2_vs, 64, 32, 96, 96);
  pack_w<<<cdiv(32 * 64, 256), 256, 0, stream>>>(W2svT, m2_sv, nullptr, 64, 0, 32, 64);
  pack_w<<<cdiv(32 * 32, 256), 256, 0, stream>>>(W2vvT, m2_vv, nullptr, 32, 0, 32, 32);
  pack_w<<<cdiv(96 * 192, 256), 256, 0, stream>>>(U1p, u1_ss, u1_vs, 128, 64, 96, 192);
  pack_w<<<cdiv(32 * 128, 256), 256, 0, stream>>>(U1svT, u1_sv, nullptr, 128, 0, 32, 128);
  pack_w<<<cdiv(32 * 64, 256), 256, 0, stream>>>(U1vvT, u1_vv, nullptr, 64, 0, 32, 64);
  pack_w<<<cdiv(64 * 96, 256), 256, 0, stream>>>(U2p, u2_ss, u2_vs, 64, 32, 64, 96);
  pack_w<<<cdiv(32 * 64, 256), 256, 0, stream>>>(U2svT, u2_sv, nullptr, 64, 0, 32, 64);
  pack_w<<<cdiv(32 * 32, 256), 256, 0, stream>>>(U2vvT, u2_vv, nullptr, 32, 0, 32, 32);

  edge_kernel<<<NEDGES / 16, 32, 0, stream>>>(
      x_s, x_v, eidx, edist, eas, eav, W1p, W1svT, W1vvT, m1_b, W2p, W2svT,
      W2vvT, m2_b, agg_s, agg_v, sum_as, sum_av, cntb);

  node_kernel<<<NNODES / 16, 32, 0, stream>>>(
      x_s, x_v, agg_s, agg_v, sum_as, sum_av, cntb, U1p, U1svT, U1vvT, u1_b,
      U2p, U2svT, U2vvT, u2_b, out);
}
